// MultiheadAttention_30459908063696
// MI455X (gfx1250) — matrix operand order repaired and checked by reading, not compiled, not run on hardware
//
#include <hip/hip_runtime.h>
#include <hip/hip_bf16.h>

// ---------------------------------------------------------------------------
// cosFormer multihead attention for MI455X (gfx1250, wave32, WMMA bf16)
// E=1024 H=16 D=64 T=S=2048 B=8  -> M=16384 rows for the big GEMMs
// ---------------------------------------------------------------------------

typedef __bf16 bf16;
typedef __attribute__((ext_vector_type(16))) __bf16 v16bf;
typedef __attribute__((ext_vector_type(8)))  __bf16 v8bf;
typedef __attribute__((ext_vector_type(4)))  __bf16 v4bf;
typedef __attribute__((ext_vector_type(8)))  float  v8f;

#define E_DIM 1024
#define H_NUM 16
#define D_DIM 64
#define T_LEN 2048
#define S_LEN 2048
#define B_NUM 8
#define BH    (B_NUM * H_NUM)   // 128
#define M_ROWS (T_LEN * B_NUM)  // 16384

// ---------------------------------------------------------------------------
// WMMA helpers: bf16 16x16x32, f32 accumulate
// ---------------------------------------------------------------------------
__device__ __forceinline__ v8f wmma_bf16(v16bf a, v16bf b, v8f c) {
  return __builtin_amdgcn_wmma_f32_16x16x32_bf16(
      /*neg_a=*/false, a, /*neg_b=*/false, b,
      /*c_mod=*/(short)0, c, /*reuse_a=*/false, /*reuse_b=*/false);
}

// A fragment, 16x32 bf16 (M x K), row m is K-contiguous in memory.
// Lane layout: lanes 0-15 hold K=[0..15]; lanes 16-31 hold K=[16..31]
// (same contiguous K order as the B fragment).
__device__ __forceinline__ v16bf load_a_frag(const bf16* __restrict__ Atile,
                                             int lda, int mOff, int kBase,
                                             int lane) {
  const int m  = mOff + (lane & 15);
  const int kh = (lane >> 4) * 16;
  const bf16* p = Atile + (size_t)m * lda + kBase + kh;
  v8bf lo = *(const v8bf*)p;          // 16B load
  v8bf hi = *(const v8bf*)(p + 8);    // 16B load
  v16bf r;
#pragma unroll
  for (int i = 0; i < 8; ++i) { r[i] = lo[i]; r[i + 8] = hi[i]; }
  return r;
}

// B fragment, 32x16 bf16 (K x N), column n is K-contiguous in memory
// (B stored as (N, K) row-major). Lanes 0-15: K=[0..15]; lanes 16-31: K=[16..31].
__device__ __forceinline__ v16bf load_b_frag(const bf16* __restrict__ Btile,
                                             int ldb, int nOff, int kBase,
                                             int lane) {
  const int n  = nOff + (lane & 15);
  const int kh = (lane >> 4) * 16;
  return *(const v16bf*)(Btile + (size_t)n * ldb + kBase + kh);  // 32B load
}

// One wave computes a 64x64 output tile: 4x4 WMMA tiles, K-loop in steps of 32.
// 16 b128 fragment loads feed 16 WMMAs per iteration (1 load : 1 wmma).
// global_prefetch_b8 pulls the panel 256 halves ahead (lanes fan over rows).
__device__ __forceinline__ void wave_gemm_64x64(v8f (&c)[4][4],
                                                const bf16* __restrict__ Atile,
                                                int lda,
                                                const bf16* __restrict__ Btile,
                                                int ldb, int K, int lane) {
  const bf16* pfA = Atile + (size_t)lane * lda;
  const bf16* pfB = Btile + (size_t)lane * ldb;
#pragma unroll 2
  for (int k = 0; k < K; k += 32) {
    if (((k & 127) == 0) && (k + 256 < K)) {
      __builtin_prefetch(pfA + k + 256, 0, 0);                      // A rows 0..31
      __builtin_prefetch(pfA + (size_t)32 * lda + k + 256, 0, 0);   // A rows 32..63
      __builtin_prefetch(pfB + k + 256, 0, 0);                      // B rows 0..31
      __builtin_prefetch(pfB + (size_t)32 * ldb + k + 256, 0, 0);   // B rows 32..63
    }
    v16bf a[4], b[4];
#pragma unroll
    for (int i = 0; i < 4; ++i) a[i] = load_a_frag(Atile, lda, i * 16, k, lane);
#pragma unroll
    for (int j = 0; j < 4; ++j) b[j] = load_b_frag(Btile, ldb, j * 16, k, lane);
#pragma unroll
    for (int i = 0; i < 4; ++i)
#pragma unroll
      for (int j = 0; j < 4; ++j) c[i][j] = wmma_bf16(a[i], b[j], c[i][j]);
  }
}

// One wave computes a 32x32 output tile (for the small per-head kv GEMM).
__device__ __forceinline__ void wave_gemm_32x32(v8f (&c)[2][2],
                                                const bf16* __restrict__ Atile,
                                                int lda,
                                                const bf16* __restrict__ Btile,
                                                int ldb, int K, int lane) {
  for (int k = 0; k < K; k += 32) {
    v16bf a0 = load_a_frag(Atile, lda, 0,  k, lane);
    v16bf a1 = load_a_frag(Atile, lda, 16, k, lane);
    v16bf b0 = load_b_frag(Btile, ldb, 0,  k, lane);
    v16bf b1 = load_b_frag(Btile, ldb, 16, k, lane);
    c[0][0] = wmma_bf16(a0, b0, c[0][0]);
    c[0][1] = wmma_bf16(a0, b1, c[0][1]);
    c[1][0] = wmma_bf16(a1, b0, c[1][0]);
    c[1][1] = wmma_bf16(a1, b1, c[1][1]);
  }
}

// ---------------------------------------------------------------------------
// f32 -> bf16 conversion (vectorized x4)
// ---------------------------------------------------------------------------
__global__ __launch_bounds__(256) void cvt_f32_bf16(
    const float* __restrict__ in, bf16* __restrict__ out, int n4) {
  int i = blockIdx.x * 256 + threadIdx.x;
  if (i < n4) {
    float4 f = ((const float4*)in)[i];
    v4bf o;
    o[0] = (bf16)f.x; o[1] = (bf16)f.y; o[2] = (bf16)f.z; o[3] = (bf16)f.w;
    ((v4bf*)out)[i] = o;
  }
}

// ---------------------------------------------------------------------------
// Projection GEMMs: Y = X @ W^T + bias, fused cosFormer epilogue.
// X: (16384, 1024) bf16 row-major (row = t*8 + b, K contiguous)
// W: (1024, 1024) bf16 row-major (row n = output feature, K contiguous)
// MODE 0: q2[bh, t, d]    = relu(scale*y)*sin ; [d+64] = *cos   (BH,T,128)
// MODE 1: k2[bh, s, d]    = relu(y)*sin      ; [d+64] = *cos    (BH,S,128)
// MODE 2: v [bh, s, d]    = y                                    (BH,S,64)
// grid: (M/256, N/128), block 256 (8 waves, 4(M) x 2(N) of 64x64 tiles)
// ---------------------------------------------------------------------------
template <int MODE>
__global__ __launch_bounds__(256) void proj_kernel(
    const bf16* __restrict__ A, const bf16* __restrict__ W,
    const float* __restrict__ bias, bf16* __restrict__ out) {
  const int lane = threadIdx.x & 31;
  const int wave = threadIdx.x >> 5;
  const int mWG = blockIdx.x * 256 + (wave & 3) * 64;
  const int nWG = blockIdx.y * 128 + (wave >> 2) * 64;

  v8f c[4][4] = {};
  wave_gemm_64x64(c, A + (size_t)mWG * E_DIM, E_DIM,
                  W + (size_t)nWG * E_DIM, E_DIM, E_DIM, lane);

  const float half_pi = 1.5707963267948966f;
#pragma unroll
  for (int i = 0; i < 4; ++i) {
#pragma unroll
    for (int j = 0; j < 4; ++j) {
      const int n = nWG + j * 16 + (lane & 15);
      const float bv = bias[n];
      const int h = n >> 6, d = n & 63;
#pragma unroll
      for (int r = 0; r < 8; ++r) {
        const int m = mWG + i * 16 + r + ((lane >> 4) << 3);
        float val = c[i][j][r] + bv;
        const int t = m >> 3, bb = m & 7;
        const int bh = bb * H_NUM + h;
        if (MODE == 2) {
          out[(size_t)bh * S_LEN * D_DIM + (size_t)t * D_DIM + d] = (bf16)val;
        } else {
          if (MODE == 0) val *= 0.125f;  // D^-0.5 = 64^-0.5
          val = fmaxf(val, 0.0f);
          float sv, cv;
          __sincosf(half_pi * (float)(t + 1) * (1.0f / 2048.0f), &sv, &cv);
          bf16* o = out + (size_t)bh * T_LEN * 128 + (size_t)t * 128;
          o[d]      = (bf16)(val * sv);
          o[d + 64] = (bf16)(val * cv);
        }
      }
    }
  }
}

// ---------------------------------------------------------------------------
// Batched bf16 transpose: per head, (R, C) -> (C, R). LDS 32x32 tiles.
// grid: (C/32, R/32, BH), block 256 (32 x 8)
// ---------------------------------------------------------------------------
__global__ __launch_bounds__(256) void transpose_bf16_kernel(
    const bf16* __restrict__ src, bf16* __restrict__ dst, int R, int C) {
  __shared__ bf16 tile[32][33];
  const size_t base = (size_t)blockIdx.z * R * C;
  const int c0 = blockIdx.x * 32, r0 = blockIdx.y * 32;
  const int tx = threadIdx.x & 31, ty = threadIdx.x >> 5;
#pragma unroll
  for (int i = 0; i < 4; ++i)
    tile[ty + i * 8][tx] = src[base + (size_t)(r0 + ty + i * 8) * C + c0 + tx];
  __syncthreads();
#pragma unroll
  for (int i = 0; i < 4; ++i)
    dst[base + (size_t)(c0 + ty + i * 8) * R + r0 + tx] = tile[tx][ty + i * 8];
}

// ---------------------------------------------------------------------------
// k2sum[bh, d] = sum_s k2t[bh, d, s]. One wave per row (rows are contiguous).
// grid: BH*128/8, block 256
// ---------------------------------------------------------------------------
__global__ __launch_bounds__(256) void rowsum_kernel(
    const bf16* __restrict__ k2t, float* __restrict__ sum) {
  const int row = blockIdx.x * 8 + (threadIdx.x >> 5);
  const int lane = threadIdx.x & 31;
  const bf16* p = k2t + (size_t)row * S_LEN;
  float acc = 0.f;
  for (int i = lane; i < S_LEN; i += 32) acc += (float)p[i];
#pragma unroll
  for (int off = 16; off > 0; off >>= 1) acc += __shfl_xor(acc, off, 32);
  if (lane == 0) sum[row] = acc;
}

// ---------------------------------------------------------------------------
// kv GEMM per head: kv = k2^T @ v.  A = k2t (128, 2048), B = vT (64, 2048)
// (both K-contiguous). Output stored transposed: kvT[bh] (64, 128) bf16.
// grid: BH, block 256 (8 waves of 32x32 covering 128x64)
// ---------------------------------------------------------------------------
__global__ __launch_bounds__(256) void kv_kernel(
    const bf16* __restrict__ k2t, const bf16* __restrict__ vT,
    bf16* __restrict__ kvT) {
  const int bh = blockIdx.x;
  const int lane = threadIdx.x & 31;
  const int wave = threadIdx.x >> 5;
  const int wm = (wave & 3) * 32;   // M in [0,128)
  const int wn = (wave >> 2) * 32;  // N in [0,64)
  const bf16* A = k2t + (size_t)bh * 128 * S_LEN + (size_t)wm * S_LEN;
  const bf16* Bm = vT + (size_t)bh * D_DIM * S_LEN + (size_t)wn * S_LEN;

  v8f c[2][2] = {};
  wave_gemm_32x32(c, A, S_LEN, Bm, S_LEN, S_LEN, lane);

  bf16* O = kvT + (size_t)bh * D_DIM * 128;
#pragma unroll
  for (int i = 0; i < 2; ++i)
#pragma unroll
    for (int j = 0; j < 2; ++j) {
      const int n = wn + j * 16 + (lane & 15);
#pragma unroll
      for (int r = 0; r < 8; ++r) {
        const int m = wm + i * 16 + r + ((lane >> 4) << 3);
        O[(size_t)n * 128 + m] = (bf16)c[i][j][r];
      }
    }
}

// ---------------------------------------------------------------------------
// num GEMM per head: num = q2 @ kv.  A = q2[bh] (2048, 128), B = kvT[bh]
// (64, 128) -> both K-contiguous.  Output f32 num[bh] (2048, 64).
// grid: (T/512, BH), block 256 (8 waves of 64x64 stacked in M)
// ---------------------------------------------------------------------------
__global__ __launch_bounds__(256) void num_kernel(
    const bf16* __restrict__ q2, const bf16* __restrict__ kvT,
    float* __restrict__ num) {
  const int bh = blockIdx.y;
  const int lane = threadIdx.x & 31;
  const int wave = threadIdx.x >> 5;
  const int mWG = blockIdx.x * 512 + wave * 64;
  const bf16* A = q2 + (size_t)bh * T_LEN * 128 + (size_t)mWG * 128;
  const bf16* Bm = kvT + (size_t)bh * D_DIM * 128;  // N = 64 fits one 64-wide tile

  v8f c[4][4] = {};
  wave_gemm_64x64(c, A, 128, Bm, 128, 128, lane);

  float* O = num + (size_t)bh * T_LEN * D_DIM;
#pragma unroll
  for (int i = 0; i < 4; ++i)
#pragma unroll
    for (int j = 0; j < 4; ++j) {
      const int n = j * 16 + (lane & 15);
#pragma unroll
      for (int r = 0; r < 8; ++r) {
        const int m = mWG + i * 16 + r + ((lane >> 4) << 3);
        O[(size_t)m * D_DIM + n] = c[i][j][r];
      }
    }
}

// ---------------------------------------------------------------------------
// finalize: den[bh,t] = dot(q2[bh,t,:], k2sum[bh,:]); out = num / max(den,eps)
// written as bf16 into attn buffer already laid out (t*8+b, h*64+d) for the
// output projection. One wave per (bh, t) row.
// grid: BH*T/8, block 256
// ---------------------------------------------------------------------------
__global__ __launch_bounds__(256) void finalize_kernel(
    const bf16* __restrict__ q2, const float* __restrict__ k2sum,
    const float* __restrict__ num, bf16* __restrict__ attn) {
  const int row = blockIdx.x * 8 + (threadIdx.x >> 5);
  const int lane = threadIdx.x & 31;
  const int bh = row >> 11;          // / 2048
  const int t = row & 2047;
  const bf16* q = q2 + (size_t)bh * T_LEN * 128 + (size_t)t * 128;
  const float* ks = k2sum + (size_t)bh * 128;
  float acc = 0.f;
#pragma unroll
  for (int i = lane; i < 128; i += 32) acc += (float)q[i] * ks[i];
#pragma unroll
  for (int off = 16; off > 0; off >>= 1) acc += __shfl_xor(acc, off, 32);
  const float inv_den = 1.0f / fmaxf(acc, 1e-6f);
  const float* nm = num + (size_t)bh * T_LEN * D_DIM + (size_t)t * D_DIM;
  const int b = bh >> 4, h = bh & 15;
  bf16* o = attn + ((size_t)t * B_NUM + b) * E_DIM + (size_t)h * D_DIM;
#pragma unroll
  for (int i = lane; i < D_DIM; i += 32) o[i] = (bf16)(nm[i] * inv_den);
}

// ---------------------------------------------------------------------------
// Output projection: d_out = attn @ wo^T + bo   (f32 output)
// grid: (M/256, N/128), block 256
// ---------------------------------------------------------------------------
__global__ __launch_bounds__(256) void out_proj_kernel(
    const bf16* __restrict__ A, const bf16* __restrict__ W,
    const float* __restrict__ bias, float* __restrict__ out) {
  const int lane = threadIdx.x & 31;
  const int wave = threadIdx.x >> 5;
  const int mWG = blockIdx.x * 256 + (wave & 3) * 64;
  const int nWG = blockIdx.y * 128 + (wave >> 2) * 64;

  v8f c[4][4] = {};
  wave_gemm_64x64(c, A + (size_t)mWG * E_DIM, E_DIM,
                  W + (size_t)nWG * E_DIM, E_DIM, E_DIM, lane);

#pragma unroll
  for (int i = 0; i < 4; ++i)
#pragma unroll
    for (int j = 0; j < 4; ++j) {
      const int n = nWG + j * 16 + (lane & 15);
      const float bv = bias[n];
#pragma unroll
      for (int r = 0; r < 8; ++r) {
        const int m = mWG + i * 16 + r + ((lane >> 4) << 3);
        out[(size_t)m * E_DIM + n] = c[i][j][r] + bv;
      }
    }
}

// ---------------------------------------------------------------------------
// Host-side orchestration
// ---------------------------------------------------------------------------
extern "C" void kernel_launch(void* const* d_in, const int* in_sizes, int n_in,
                              void* d_out, int out_size, void* d_ws,
                              size_t ws_size, hipStream_t stream) {
  const float* query  = (const float*)d_in[0];
  const float* key_in = (const float*)d_in[1];
  const float* value  = (const float*)d_in[2];
  const float* wq = (const float*)d_in[3];
  const float* bq = (const float*)d_in[4];
  const float* wk = (const float*)d_in[5];
  const float* bk = (const float*)d_in[6];
  const float* wv = (const float*)d_in[7];
  const float* bv = (const float*)d_in[8];
  const float* wo = (const float*)d_in[9];
  const float* bo = (const float*)d_in[10];
  float* outp = (float*)d_out;

  // ---- workspace layout (bytes, 256B aligned blocks) ----
  char* ws = (char*)d_ws;
  size_t off = 0;
  auto alloc = [&](size_t bytes) {
    size_t o = off;
    off += (bytes + 255) & ~(size_t)255;
    return o;
  };
  const size_t W_ELEMS = (size_t)E_DIM * E_DIM;            // 1M
  const size_t X_ELEMS = (size_t)M_ROWS * E_DIM;           // 16M
  size_t o_wbf  = alloc(4 * W_ELEMS * 2);                  // 8  MB: wq,wk,wv,wo bf16
  size_t o_qb   = alloc(X_ELEMS * 2);                      // 32 MB
  size_t o_kb   = alloc(X_ELEMS * 2);                      // 32 MB (qb+kb reused as num)
  size_t o_vb   = alloc(X_ELEMS * 2);                      // 32 MB (reused as attn)
  size_t o_q2   = alloc((size_t)BH * T_LEN * 128 * 2);     // 64 MB
  size_t o_k2   = alloc((size_t)BH * S_LEN * 128 * 2);     // 64 MB
  size_t o_v    = alloc((size_t)BH * S_LEN * D_DIM * 2);   // 32 MB
  size_t o_k2t  = alloc((size_t)BH * 128 * S_LEN * 2);     // 64 MB
  size_t o_vT   = alloc((size_t)BH * D_DIM * S_LEN * 2);   // 32 MB
  size_t o_kvT  = alloc((size_t)BH * D_DIM * 128 * 2);     // 2  MB
  size_t o_ks   = alloc((size_t)BH * 128 * 4);             // 64 KB
  size_t o_num  = o_qb;  // 64 MB f32, reuses qb+kb (dead after projections)
  size_t o_attn = o_vb;  // 32 MB bf16, reuses vb   (dead after V projection)
  (void)ws_size; (void)in_sizes; (void)n_in; (void)out_size;

  bf16* wbf  = (bf16*)(ws + o_wbf);
  bf16* wqb = wbf;
  bf16* wkb = wbf + W_ELEMS;
  bf16* wvb = wbf + 2 * W_ELEMS;
  bf16* wob = wbf + 3 * W_ELEMS;
  bf16* qb   = (bf16*)(ws + o_qb);
  bf16* kb   = (bf16*)(ws + o_kb);
  bf16* vb   = (bf16*)(ws + o_vb);
  bf16* q2   = (bf16*)(ws + o_q2);
  bf16* k2   = (bf16*)(ws + o_k2);
  bf16* vh   = (bf16*)(ws + o_v);
  bf16* k2t  = (bf16*)(ws + o_k2t);
  bf16* vT   = (bf16*)(ws + o_vT);
  bf16* kvT  = (bf16*)(ws + o_kvT);
  float* k2s = (float*)(ws + o_ks);
  float* num = (float*)(ws + o_num);
  bf16* attn = (bf16*)(ws + o_attn);

  // 1) f32 -> bf16 conversions
  {
    int n4x = (int)(X_ELEMS / 4);   // 4M
    int n4w = (int)(W_ELEMS / 4);   // 256K
    cvt_f32_bf16<<<(n4x + 255) / 256, 256, 0, stream>>>(query,  qb,  n4x);
    cvt_f32_bf16<<<(n4x + 255) / 256, 256, 0, stream>>>(key_in, kb,  n4x);
    cvt_f32_bf16<<<(n4x + 255) / 256, 256, 0, stream>>>(value,  vb,  n4x);
    cvt_f32_bf16<<<(n4w + 255) / 256, 256, 0, stream>>>(wq, wqb, n4w);
    cvt_f32_bf16<<<(n4w + 255) / 256, 256, 0, stream>>>(wk, wkb, n4w);
    cvt_f32_bf16<<<(n4w + 255) / 256, 256, 0, stream>>>(wv, wvb, n4w);
    cvt_f32_bf16<<<(n4w + 255) / 256, 256, 0, stream>>>(wo, wob, n4w);
  }

  // 2) projections with fused cosFormer epilogue
  dim3 gProj(M_ROWS / 256, E_DIM / 128);
  proj_kernel<0><<<gProj, 256, 0, stream>>>(qb, wqb, bq, q2);
  proj_kernel<1><<<gProj, 256, 0, stream>>>(kb, wkb, bk, k2);
  proj_kernel<2><<<gProj, 256, 0, stream>>>(vb, wvb, bv, vh);

  // 3) transposes for K-contiguous WMMA fragments in the kv GEMM
  transpose_bf16_kernel<<<dim3(128 / 32, S_LEN / 32, BH), 256, 0, stream>>>(
      k2, k2t, S_LEN, 128);
  transpose_bf16_kernel<<<dim3(D_DIM / 32, S_LEN / 32, BH), 256, 0, stream>>>(
      vh, vT, S_LEN, D_DIM);

  // 4) k2sum (per head feature sums)
  rowsum_kernel<<<(BH * 128) / 8, 256, 0, stream>>>(k2t, k2s);

  // 5) kv = k2^T @ v (per head), stored transposed
  kv_kernel<<<BH, 256, 0, stream>>>(k2t, vT, kvT);

  // 6) num = q2 @ kv (per head)
  num_kernel<<<dim3(T_LEN / 512, BH), 256, 0, stream>>>(q2, kvT, num);

  // 7) den + divide + permute into output-projection layout
  finalize_kernel<<<(BH * T_LEN) / 8, 256, 0, stream>>>(q2, k2s, num, attn);

  // 8) output projection -> d_out (f32)
  out_proj_kernel<<<dim3(M_ROWS / 256, E_DIM / 128), 256, 0, stream>>>(
      attn, wob, bo, outp);
}
